// CausalSelfAttention_29858612641874
// MI455X (gfx1250) — matrix operand order repaired and checked by reading, not compiled, not run on hardware
//
#include <hip/hip_runtime.h>

typedef unsigned short u16;
typedef __attribute__((ext_vector_type(16))) __bf16 v16bf;
typedef __attribute__((ext_vector_type(8)))  float  v8f;
typedef __attribute__((ext_vector_type(4)))  unsigned int u32x4;
typedef __attribute__((ext_vector_type(8)))  int i32x8;
typedef __attribute__((ext_vector_type(4)))  int i32x4;

constexpr int Bc = 4, Tc = 1024, Ec = 1024, Hc = 16, Dc = 64;
constexpr int QKV_LD = 3 * Ec;   // qkv row stride
constexpr int VSTR = 40;         // padded LDS stride (u16): keeps 16B align, avoids bank conflicts

__device__ __forceinline__ u16 f2bf(float x) {
  union { __bf16 b; u16 u; } c;
  c.b = (__bf16)x;               // native cvt (RTE)
  return c.u;
}

union FragB {
  uint4 q[2];
  v16bf v;
  u16   h[16];
};

// ---------------------------------------------------------------------------
// f32 -> bf16 elementwise convert (weights)
// ---------------------------------------------------------------------------
__global__ __launch_bounds__(256) void k_cvt_bf16(const float* __restrict__ s,
                                                  u16* __restrict__ d, int n) {
  int i = (blockIdx.x * 256 + threadIdx.x) * 8;
  if (i + 8 > n) return;
  float4 a = *(const float4*)(s + i);
  float4 b = *(const float4*)(s + i + 4);
  uint4 pk;
  pk.x = f2bf(a.x) | ((unsigned)f2bf(a.y) << 16);
  pk.y = f2bf(a.z) | ((unsigned)f2bf(a.w) << 16);
  pk.z = f2bf(b.x) | ((unsigned)f2bf(b.y) << 16);
  pk.w = f2bf(b.z) | ((unsigned)f2bf(b.w) << 16);
  *(uint4*)(d + i) = pk;
}

// ---------------------------------------------------------------------------
// RMSNorm (one row of E=1024 per 256-thread block) -> bf16
// ---------------------------------------------------------------------------
__global__ __launch_bounds__(256) void k_rmsnorm(const float* __restrict__ x,
                                                 const float* __restrict__ scale,
                                                 u16* __restrict__ xn) {
  const int row = blockIdx.x;
  const int tid = threadIdx.x;
  const float4 v = *(const float4*)(x + (long)row * Ec + tid * 4);
  float ss = v.x * v.x + v.y * v.y + v.z * v.z + v.w * v.w;
#pragma unroll
  for (int off = 16; off > 0; off >>= 1) ss += __shfl_xor(ss, off, 32);
  __shared__ float red[8];
  if ((tid & 31) == 0) red[tid >> 5] = ss;
  __syncthreads();
  float tot = 0.f;
#pragma unroll
  for (int i = 0; i < 8; ++i) tot += red[i];
  const float inv = rsqrtf(tot * (1.0f / Ec) + 1e-5f);
  const float4 sc = *(const float4*)(scale + tid * 4);
  uint2 pk;
  pk.x = f2bf(v.x * inv * sc.x) | ((unsigned)f2bf(v.y * inv * sc.y) << 16);
  pk.y = f2bf(v.z * inv * sc.z) | ((unsigned)f2bf(v.w * inv * sc.w) << 16);
  *(uint2*)(xn + (long)row * Ec + tid * 4) = pk;
}

// ---------------------------------------------------------------------------
// GEMM: C[M,N] = A[M,K] (bf16 row-major) x W[N,K]^T (bf16 torch-Linear weight)
// One wave = 32x64 C tile (2x4 WMMA frags). Software pipelined with TWO NAMED
// buffer sets and a 2x-unrolled K loop (K % 64 == 0) so every fragment array
// is indexed by compile-time constants -> SROA keeps them in VGPRs (the
// runtime `buf` index in rounds 4-5 forced the buffers into scratch).
// ---------------------------------------------------------------------------
template <bool BF16OUT>
__global__ __launch_bounds__(128, 1) void k_gemm(const u16* __restrict__ A,
                                                 const u16* __restrict__ W,
                                                 void* __restrict__ Cout,
                                                 int M, int N, int K) {
  const int lane = threadIdx.x & 31;
  const int wid  = threadIdx.x >> 5;
  const int hi = lane >> 4, ln = lane & 15;
  const int m0 = blockIdx.y * 64 + (wid >> 1) * 32;
  const int n0 = blockIdx.x * 128 + (wid & 1) * 64;

  v8f c[2][4] = {};
  const u16* arow[2];
  arow[0] = A + (long)(m0 + ln) * K;
  arow[1] = arow[0] + (long)16 * K;
  const u16* wrow[4];
#pragma unroll
  for (int nt = 0; nt < 4; ++nt) wrow[nt] = W + (long)(n0 + nt * 16 + ln) * K;

  FragB aA[2], bA[4], aB[2], bB[4];
  auto ld = [&](FragB (&aT)[2], FragB (&bT)[4], int k0) {
#pragma unroll
    for (int mt = 0; mt < 2; ++mt) {
      // A-frag: halves 0..7 -> K = hi*8 + j ; halves 8..15 -> K = 16 + hi*8 + j
      aT[mt].q[0] = *(const uint4*)(arow[mt] + k0 + hi * 8);
      aT[mt].q[1] = *(const uint4*)(arow[mt] + k0 + 16 + hi * 8);
    }
#pragma unroll
    for (int nt = 0; nt < 4; ++nt) {
      // B-frag: lane = col n (weight row); halves 0..7 -> K = hi*8 + j ; halves 8..15 -> K = 16 + hi*8 + j
      bT[nt].q[0] = *(const uint4*)(wrow[nt] + k0 + hi * 8);
      bT[nt].q[1] = *(const uint4*)(wrow[nt] + k0 + 16 + hi * 8);
    }
  };
  auto mm = [&](FragB (&aT)[2], FragB (&bT)[4]) {
#pragma unroll
    for (int mt = 0; mt < 2; ++mt)
#pragma unroll
      for (int nt = 0; nt < 4; ++nt)
        c[mt][nt] = __builtin_amdgcn_wmma_f32_16x16x32_bf16(
            false, aT[mt].v, false, bT[nt].v, (short)0, c[mt][nt], false, false);
  };

  ld(aA, bA, 0);
  for (int k0 = 0; k0 < K; k0 += 64) {           // K multiple of 64
    ld(aB, bB, k0 + 32);                         // B-buffer loads in flight
    if (k0 + 512 < K) {
      __builtin_prefetch(arow[0] + k0 + 512, 0, 0);  // global_prefetch_b8 into L2
      __builtin_prefetch(wrow[0] + k0 + 512, 0, 0);
    }
    mm(aA, bA);
    if (k0 + 64 < K) ld(aA, bA, k0 + 64);        // refill A while B computes
    mm(aB, bB);
  }

#pragma unroll
  for (int mt = 0; mt < 2; ++mt)
#pragma unroll
    for (int nt = 0; nt < 4; ++nt)
#pragma unroll
      for (int v = 0; v < 8; ++v) {
        const long m = m0 + mt * 16 + v + hi * 8;
        const long n = n0 + nt * 16 + ln;
        if (BF16OUT) ((u16*)Cout)[m * N + n] = f2bf(c[mt][nt][v]);
        else         ((float*)Cout)[m * N + n] = c[mt][nt][v];
      }
}

// ---------------------------------------------------------------------------
// Flash attention, one wave per (b, h, 16-query tile).
// S^T = K·Q^T so the C-layout of exp(S^T) equals the A-layout of P for the
// P·V WMMA. V staged transposed ([d][key]) in LDS. Pipelining uses TWO NAMED
// K/V buffer sets with a 2x-unrolled key loop (constant indices only -> no
// scratch spills). The current V buffer is drained into LDS at the top of
// each step (loaded one step earlier -> no loadcnt wait on the stores).
// ---------------------------------------------------------------------------
__global__ __launch_bounds__(128, 1) void k_attn(const u16* __restrict__ qkv,
                                                 u16* __restrict__ out) {
  __shared__ u16 vlds[4 * Dc * VSTR];
  const int lane = threadIdx.x & 31;
  const int wid  = threadIdx.x >> 5;
  const int w  = blockIdx.x * 4 + wid;       // 0 .. B*H*64-1
  const int qt = w & 63;
  const int h  = (w >> 6) & 15;
  const int b  = w >> 10;
  const int hi = lane >> 4, ln = lane & 15;
  u16* vt = vlds + wid * Dc * VSTR;

  // Q^T B-frags (lane column = query ln), held for entire kernel
  FragB bq[2];
  {
    const u16* qp = qkv + (long)(b * Tc + qt * 16 + ln) * QKV_LD + h * Dc;
#pragma unroll
    for (int kk = 0; kk < 2; ++kk) {
      bq[kk].q[0] = *(const uint4*)(qp + kk * 32 + hi * 8);
      bq[kk].q[1] = *(const uint4*)(qp + kk * 32 + 16 + hi * 8);
    }
  }

  const u16* kbase = qkv + (long)(b * Tc) * QKV_LD + Ec + h * Dc;
  const u16* vbase = qkv + (long)(b * Tc) * QKV_LD + 2 * Ec + h * Dc;

  v8f o[4] = {};
  float m_run = -__builtin_inff();
  float l_run = 0.f;
  const int qg = qt * 16 + ln;          // this lane's query index
  const float sscale = 0.125f;          // 1/sqrt(64)
  const int kend = (qt + 1) * 16;

  FragB akA[2][2], akB[2][2];           // [key subtile][kk]
  uint4 vA[8], vB[8];

  auto load_kv = [&](FragB (&ak)[2][2], uint4 (&vv)[8], int k0) {
    const u16* kp0 = kbase + (long)(k0 + ln) * QKV_LD;
    const u16* kp1 = kp0 + (long)16 * QKV_LD;
#pragma unroll
    for (int kk = 0; kk < 2; ++kk) {
      ak[0][kk].q[0] = *(const uint4*)(kp0 + kk * 32 + hi * 8);
      ak[0][kk].q[1] = *(const uint4*)(kp0 + kk * 32 + 16 + hi * 8);
      ak[1][kk].q[0] = *(const uint4*)(kp1 + kk * 32 + hi * 8);
      ak[1][kk].q[1] = *(const uint4*)(kp1 + kk * 32 + 16 + hi * 8);
    }
    const u16* vp = vbase + (long)(k0 + lane) * QKV_LD;
#pragma unroll
    for (int c8 = 0; c8 < 8; ++c8) vv[c8] = *(const uint4*)(vp + c8 * 8);
  };

  auto step = [&](FragB (&ak)[2][2], uint4 (&vv)[8], int k0) {
    // ---- drain V buffer into LDS (loaded last step; no wait): vt[d*VSTR+key]
#pragma unroll
    for (int c8 = 0; c8 < 8; ++c8) {
      union { uint4 u; u16 hh[8]; } t;
      t.u = vv[c8];
#pragma unroll
      for (int j = 0; j < 8; ++j) vt[(c8 * 8 + j) * VSTR + lane] = t.hh[j];
    }

    // ---- S^T = K · Q^T : two 16-key subtiles, D=64 -> two k=32 steps ----
    v8f c0 = {}, c1 = {};
#pragma unroll
    for (int kk = 0; kk < 2; ++kk) {
      c0 = __builtin_amdgcn_wmma_f32_16x16x32_bf16(false, ak[0][kk].v, false, bq[kk].v, (short)0, c0, false, false);
      c1 = __builtin_amdgcn_wmma_f32_16x16x32_bf16(false, ak[1][kk].v, false, bq[kk].v, (short)0, c1, false, false);
    }

    // ---- causal mask + online softmax (lane column = query) ----
    float mnew = m_run;
#pragma unroll
    for (int v = 0; v < 8; ++v) {
      const int kg = k0 + v + hi * 8;   // key of c0[v]; c1[v] is kg+16
      float s0 = c0[v] * sscale;
      float s1 = c1[v] * sscale;
      if (kg > qg)      s0 = -__builtin_inff();
      if (kg + 16 > qg) s1 = -__builtin_inff();
      c0[v] = s0; c1[v] = s1;
      mnew = fmaxf(mnew, fmaxf(s0, s1));
    }
    mnew = fmaxf(mnew, __shfl_xor(mnew, 16, 32));
    const float alpha = __expf(m_run - mnew);
    m_run = mnew;

    FragB pA;                // P in A-layout, directly from C-layout of S^T
    float rsum = 0.f;
#pragma unroll
    for (int v = 0; v < 8; ++v) {
      const float p0 = __expf(c0[v] - mnew);
      const float p1 = __expf(c1[v] - mnew);
      rsum += p0 + p1;
      pA.h[v]     = f2bf(p0);
      pA.h[v + 8] = f2bf(p1);
    }
    rsum += __shfl_xor(rsum, 16, 32);
    l_run = l_run * alpha + rsum;

    // rescale O accumulators (O row = query = v + hi*8; stats live in lane q)
#pragma unroll
    for (int v = 0; v < 8; ++v) {
      const float av = __shfl(alpha, v + hi * 8, 32);
#pragma unroll
      for (int dt = 0; dt < 4; ++dt) o[dt][v] *= av;
    }

    asm volatile("s_wait_dscnt 0" ::: "memory");  // V^T stores visible before B-frag loads

    // ---- O += P · V (B-frag: lane col = d, keys hi*8+j and 16+hi*8+j from LDS) ----
#pragma unroll
    for (int dt = 0; dt < 4; ++dt) {
      FragB bv;
      const u16* lp = vt + (dt * 16 + ln) * VSTR + hi * 8;
      bv.q[0] = *(const uint4*)(lp);
      bv.q[1] = *(const uint4*)(lp + 16);
      o[dt] = __builtin_amdgcn_wmma_f32_16x16x32_bf16(false, pA.v, false, bv.v, (short)0, o[dt], false, false);
    }
  };

  load_kv(akA, vA, 0);
  for (int k0 = 0; k0 < kend; k0 += 64) {
    const bool haveB = (k0 + 32 < kend);
    if (haveB) load_kv(akB, vB, k0 + 32);        // B loads in flight over step A
    step(akA, vA, k0);
    if (haveB) {
      if (k0 + 64 < kend) load_kv(akA, vA, k0 + 64);  // refill A over step B
      step(akB, vB, k0 + 32);
    }
  }

  // ---- finalize: divide by l, write bf16 [B*T, E] ----
  const float linv = 1.f / l_run;
#pragma unroll
  for (int v = 0; v < 8; ++v) {
    const float lv = __shfl(linv, v + hi * 8, 32);
    const long row = (long)b * Tc + qt * 16 + v + hi * 8;
#pragma unroll
    for (int dt = 0; dt < 4; ++dt)
      out[row * Ec + h * Dc + dt * 16 + ln] = f2bf(o[dt][v] * lv);
  }
}

// ---------------------------------------------------------------------------
// TDM probe: stage a 64x64 f32 tile of w_proj into LDS with the Tensor Data
// Mover, wait on TENSORcnt, copy to scratch (later fully overwritten by the
// QKV GEMM -> output-neutral, deterministic). D# built per ISA 08 §8.3-8.4.
// clang-23 6-arg builtin: (u32x4 g0, i32x8 g1, i32x4 g2, i32x4 g3, i32x8, i32 cpol)
// ---------------------------------------------------------------------------
__global__ __launch_bounds__(256) void k_tdm_stage(const float* __restrict__ src,
                                                   float* __restrict__ dst) {
#if __has_builtin(__builtin_amdgcn_tensor_load_to_lds)
  __shared__ float tile[64 * 64];
  if (threadIdx.x < 32) {                       // one wave issues the DMA
    const unsigned long long ga = (unsigned long long)(const void*)src;
    u32x4 g0;
    g0[0] = 1u;                                 // count=1, is_restore=0, gather=0
    g0[1] = 0u;                                 // lds_addr = 0 (tile is at LDS offset 0)
    g0[2] = (unsigned)ga;                       // global_addr[31:0]
    g0[3] = (unsigned)(ga >> 32) | (2u << 30);  // global_addr[56:32] | type=2 (image)
    i32x8 g1;
    g1[0] = (2 << 16);                          // wg_mask=0, data_size=2 (4 bytes)
    g1[1] = (64 << 16);                         // tensor_dim0 = 64  (bits 79:48)
    g1[2] = (64 << 16);                         // tensor_dim1 = 64  (bits 111:80)
    g1[3] = (64 << 16);                         // tile_dim0   = 64  (bits 127:112)
    g1[4] = 64;                                 // tile_dim1   = 64  (bits 143:128)
    g1[5] = 1024;                               // tensor_dim0_stride = 1024 (bits 207:160)
    g1[6] = 0;
    g1[7] = 0;
    i32x4 z4 = {0, 0, 0, 0};
    i32x8 z8 = {0, 0, 0, 0, 0, 0, 0, 0};
    __builtin_amdgcn_tensor_load_to_lds(g0, g1, z4, z4, z8, 0);
    __builtin_amdgcn_s_wait_tensorcnt(0);
  }
  __syncthreads();
#pragma unroll
  for (int i = 0; i < 16; ++i)
    dst[threadIdx.x + 256 * i] = tile[threadIdx.x + 256 * i];
#endif
}

// ---------------------------------------------------------------------------
extern "C" void kernel_launch(void* const* d_in, const int* in_sizes, int n_in,
                              void* d_out, int out_size, void* d_ws, size_t ws_size,
                              hipStream_t stream) {
  const float* x      = (const float*)d_in[0];
  const float* scale  = (const float*)d_in[1];
  const float* w_qkv  = (const float*)d_in[2];
  const float* w_proj = (const float*)d_in[3];
  float* out = (float*)d_out;

  // workspace layout (40 MB total)
  char* ws = (char*)d_ws;
  u16* xn    = (u16*)(ws);                      //  8 MB: RMSNorm out, reused as attn out
  u16* wqkvb = (u16*)(ws + (8u  << 20));        //  6 MB
  u16* wprjb = (u16*)(ws + (14u << 20));        //  2 MB
  u16* qkvb  = (u16*)(ws + (16u << 20));        // 24 MB
  u16* attno = xn;                              // alias: xn dead after QKV GEMM

  k_tdm_stage<<<1, 256, 0, stream>>>(w_proj, (float*)qkvb);  // scratch, overwritten below

  k_cvt_bf16<<<(3 * Ec * Ec) / 2048, 256, 0, stream>>>(w_qkv, wqkvb, 3 * Ec * Ec);
  k_cvt_bf16<<<(Ec * Ec) / 2048, 256, 0, stream>>>(w_proj, wprjb, Ec * Ec);
  k_rmsnorm<<<Bc * Tc, 256, 0, stream>>>(x, scale, xn);

  dim3 gqkv(3 * Ec / 128, Bc * Tc / 64);
  k_gemm<true><<<gqkv, 128, 0, stream>>>(xn, wqkvb, (void*)qkvb, Bc * Tc, 3 * Ec, Ec);

  k_attn<<<(Bc * Hc * (Tc / 16)) / 4, 128, 0, stream>>>(qkvb, attno);

  dim3 gprj(Ec / 128, Bc * Tc / 64);
  k_gemm<false><<<gprj, 128, 0, stream>>>(attno, wprjb, (void*)out, Bc * Tc, Ec, Ec);
}
